// SGSA_36490042147634
// MI455X (gfx1250) — matrix operand order repaired and checked by reading, not compiled, not run on hardware
//
#include <hip/hip_runtime.h>

// ---------------------------------------------------------------------------
// Types / helpers
// ---------------------------------------------------------------------------
typedef __attribute__((ext_vector_type(16))) __bf16 bf16x16;
typedef __attribute__((ext_vector_type(8)))  float  f32x8;

union Frag16 { unsigned short u[16]; bf16x16 v; };

// native f32 -> bf16 convert (v_cvt_*_bf16_f32), bit-pattern out
__device__ __forceinline__ unsigned short f2bf(float f) {
    union { __bf16 b; unsigned short u; } v;
    v.b = (__bf16)f;
    return v.u;
}

__device__ __forceinline__ float sigm(float x) { return 1.0f / (1.0f + __expf(-x)); }

__device__ __forceinline__ float waveRed(float v) {
    #pragma unroll
    for (int o = 16; o > 0; o >>= 1) v += __shfl_down(v, o, 32);
    return v;
}

#define BB 4
#define HH 256
#define WW 256
#define HWC (HH * WW)

// ---------------------------------------------------------------------------
// Generic bf16-WMMA GEMM:  Y[b, m, p] = sum_k W[m,k] * X[b, k, p]  (+bias, +resid)
//   W: [M,K] row-major f32 (optionally per-batch via wstride_b, used for Weff)
//   X rows 0..KA-1 from Xa, rows KA..K-1 from Xb (both [B, ., HW] f32)
//   Y rows 0..MA-1 -> Ya, rows MA..M-1 -> Yb
//
// 256 threads = 8 waves.  Per block: one PX-pixel strip of one batch, all M
// tiles.  Weights staged once to LDS in bf16.  Activation K-tiles are copied
// raw (f32) into LDS with GLOBAL_LOAD_ASYNC_TO_LDS_B128 (ASYNCcnt), then
// transposed/converted LDS->LDS into the bf16 fragment buffer; the next
// tile's async copy overlaps the current tile's WMMA chain.
// ---------------------------------------------------------------------------
template <int M, int K, int PX, int SUBS>
__global__ __launch_bounds__(256)
void gemm_bf16_kernel(const float* __restrict__ W, size_t wstride_b,
                      const float* __restrict__ Xa, const float* __restrict__ Xb, int KA,
                      const float* __restrict__ bias,
                      const float* __restrict__ resid,
                      float* __restrict__ Ya, float* __restrict__ Yb, int MA)
{
    constexpr int MT   = M / 16;
    constexpr int NPT  = PX / 16;
    constexpr int NSUB = MT * NPT;
    static_assert(NSUB == 8 * SUBS, "each wave must own exactly SUBS subtiles");

    __shared__ alignas(16) unsigned short Wlds[M * K];   // bf16 weights
    __shared__ alignas(16) unsigned short Bbf[PX * 32];  // bf16 transposed B frags
    __shared__ alignas(16) float          Braw[32 * PX]; // raw async-landed f32 tile

    const int tid  = threadIdx.x;
    const int lane = tid & 31;
    const int wave = tid >> 5;
    const int b    = blockIdx.y;
    const int s0   = blockIdx.x * PX;

    const float* XA = Xa + (size_t)b * (size_t)KA * HWC;
    const float* XB = Xb ? (Xb + (size_t)b * (size_t)(K - KA) * HWC) : nullptr;

    // Issue the async LDS copy of one 32 x PX f32 K-tile (16B per chunk).
    auto issue_async = [&](int k0) {
        constexpr int CHUNKS = PX / 32;          // 16B chunks per thread
        #pragma unroll
        for (int j = 0; j < CHUNKS; ++j) {
            const int idx = tid + j * 256;
            const int fo  = idx * 4;             // float offset in tile = k*PX + col
            const int k   = fo / PX;
            const int col = fo % PX;
            const int kk  = k0 + k;
            const float* src = (kk < KA) ? (XA + (size_t)kk * HWC)
                                         : (XB + (size_t)(kk - KA) * HWC);
            const float* g = src + s0 + col;
            const unsigned ldsOff = (unsigned)(uintptr_t)(&Braw[fo]);
            asm volatile("global_load_async_to_lds_b128 %0, %1, off"
                         :: "v"(ldsOff), "v"(g) : "memory");
        }
    };

    issue_async(0);  // overlap first tile copy with weight staging

    // Stage weights (per-batch when wstride_b != 0) to LDS as bf16.
    const float* Wb = W + (size_t)b * wstride_b;
    for (int i = tid * 2; i < M * K; i += 512) {
        Wlds[i]     = f2bf(Wb[i]);
        Wlds[i + 1] = f2bf(Wb[i + 1]);
    }

    f32x8 acc[SUBS] = {};

    asm volatile("s_wait_asynccnt 0x0" ::: "memory");
    __syncthreads();   // Braw tile 0 + Wlds visible to all waves

    for (int k0 = 0; k0 < K; k0 += 32) {
        // Transpose/convert Braw (f32, [k][col]) -> Bbf (bf16, [col][k]).
        {
            constexpr int CPT = PX / 8;          // columns per thread
            const int k  = tid & 31;
            const int c0 = (tid >> 5) * CPT;
            #pragma unroll
            for (int i = 0; i < CPT; ++i)
                Bbf[(c0 + i) * 32 + k] = f2bf(Braw[k * PX + c0 + i]);
        }
        __syncthreads();                         // Bbf ready; Braw drained
        if (k0 + 32 < K) issue_async(k0 + 32);   // prefetch next tile during WMMA

        // Prefetch all fragments, then issue the WMMA chain back-to-back.
        Frag16 afr[SUBS], bfr[SUBS];
        #pragma unroll
        for (int u = 0; u < SUBS; ++u) {
            const int t  = wave + u * 8;
            const int mt = t % MT;
            const int pt = t / MT;
            // A (16x32 bf16): lanes 0-15 K runs {0..7,16..23}, lanes 16-31 {8..15,24..31}
            const int abase = (mt * 16 + (lane & 15)) * K + k0 + ((lane & 16) ? 8 : 0);
            #pragma unroll
            for (int e = 0; e < 8; ++e) {
                afr[u].u[e]     = Wlds[abase + e];
                afr[u].u[8 + e] = Wlds[abase + 16 + e];
            }
            // B (32x16 bf16): lanes 0-15 K runs {0..7,16..23}, lanes 16-31 {8..15,24..31}, col=lane&15
            const int bbase = (pt * 16 + (lane & 15)) * 32 + ((lane & 16) ? 8 : 0);
            #pragma unroll
            for (int e = 0; e < 16; ++e) bfr[u].u[e] = Bbf[bbase + e + ((e & 8) ? 8 : 0)];
        }
        #pragma unroll
        for (int u = 0; u < SUBS; ++u)
            acc[u] = __builtin_amdgcn_wmma_f32_16x16x32_bf16(
                false, afr[u].v, false, bfr[u].v, (short)0, acc[u], false, false);

        if (k0 + 32 < K) {
            asm volatile("s_wait_asynccnt 0x0" ::: "memory");
            __syncthreads();                     // next Braw tile fully landed
        }
    }

    // Store: C/D layout -> lane<16: N=lane, M=row r; lane>=16: N=lane-16, M=8+r.
    #pragma unroll
    for (int u = 0; u < SUBS; ++u) {
        const int t     = wave + u * 8;
        const int mt    = t % MT;
        const int pt    = t / MT;
        const int col   = s0 + pt * 16 + (lane & 15);
        const int mrow0 = mt * 16 + ((lane & 16) ? 8 : 0);
        #pragma unroll
        for (int r = 0; r < 8; ++r) {
            const int o = mrow0 + r;
            float y = acc[u][r];
            if (bias)  y += bias[o];
            if (resid) y += resid[((size_t)b * M + o) * HWC + col];
            if (o < MA) Ya[((size_t)b * MA + o) * HWC + col] = y;
            else        Yb[((size_t)b * (M - MA) + (o - MA)) * HWC + col] = y;
        }
    }
}

// ---------------------------------------------------------------------------
// Small VALU kernels
// ---------------------------------------------------------------------------
__global__ void zero_stats_kernel(float* __restrict__ s) {
    const int i = blockIdx.x * 256 + threadIdx.x;
    if (i < 4096) s[i] = 0.0f;
}

// per-(b,c) mean-pool numerator over HW (one block each, deterministic tree)
__global__ __launch_bounds__(256)
void pool_kernel(const float* __restrict__ x1, float* __restrict__ pooled) {
    const int c = blockIdx.x, b = blockIdx.y;
    const float* p = x1 + ((size_t)b * 64 + c) * HWC;
    float s = 0.0f;
    for (int i = threadIdx.x; i < HWC; i += 256) s += p[i];
    __shared__ float red[256];
    red[threadIdx.x] = s; __syncthreads();
    for (int st = 128; st > 0; st >>= 1) {
        if (threadIdx.x < st) red[threadIdx.x] += red[threadIdx.x + st];
        __syncthreads();
    }
    if (threadIdx.x == 0) pooled[b * 64 + c] = red[0];
}

// tiny per-group channel-attention MLP -> cw[b,64]
__global__ void mid_kernel(const float* __restrict__ pooled,
                           const float* __restrict__ w1, const float* __restrict__ b1,
                           const float* __restrict__ w2, const float* __restrict__ b2,
                           float* __restrict__ cw, float invHW)
{
    const int t = threadIdx.x;
    const int b = t >> 6, idx = t & 63, g = idx >> 3, c = idx & 7;
    float hid[2];
    #pragma unroll
    for (int h = 0; h < 2; ++h) {
        float a = b1[g * 2 + h];
        #pragma unroll
        for (int cc = 0; cc < 8; ++cc)
            a += pooled[b * 64 + g * 8 + cc] * invHW * w1[g * 16 + h * 8 + cc];
        hid[h] = a > 0.0f ? a : 0.0f;
    }
    float a = b2[g * 8 + c];
    #pragma unroll
    for (int h = 0; h < 2; ++h) a += hid[h] * w2[g * 16 + c * 2 + h];
    cw[b * 64 + idx] = sigm(a);
}

// pass 1: accumulate sum of resp per (b,g) for the threshold
__global__ __launch_bounds__(256)
void cbam1_kernel(const float* __restrict__ x1, const float* __restrict__ cw,
                  const float* __restrict__ sw, const float* __restrict__ sb,
                  float* __restrict__ rsum)
{
    const int b = blockIdx.y;
    const int s = blockIdx.x * 256 + threadIdx.x;
    const float* xp = x1 + (size_t)b * 64 * HWC + s;
    const int lane = threadIdx.x & 31;
    #pragma unroll
    for (int g = 0; g < 8; ++g) {
        float xc[8]; float dot = sb[g];
        #pragma unroll
        for (int c = 0; c < 8; ++c) {
            float v = xp[(size_t)(g * 8 + c) * HWC] * cw[b * 64 + g * 8 + c];
            xc[c] = v; dot += sw[g * 8 + c] * v;
        }
        const float sp = sigm(dot);
        float acc = 0.0f;
        #pragma unroll
        for (int c = 0; c < 8; ++c) acc += sigm(xc[c] * sp);
        const float r = waveRed(acc);
        if (lane == 0) atomicAdd(&rsum[b * 8 + g], r);
    }
}

// pass 2: apply threshold mask, write masked activations
__global__ __launch_bounds__(256)
void mask_kernel(const float* __restrict__ x1, const float* __restrict__ cw,
                 const float* __restrict__ sw, const float* __restrict__ sb,
                 const float* __restrict__ rsum, float* __restrict__ xm)
{
    const int b = blockIdx.y;
    const int s = blockIdx.x * 256 + threadIdx.x;
    const float* xp = x1 + (size_t)b * 64 * HWC + s;
    float* op = xm + (size_t)b * 64 * HWC + s;
    const float inv = 1.0f / (8.0f * (float)HWC);
    #pragma unroll
    for (int g = 0; g < 8; ++g) {
        float xr[8], xc[8]; float dot = sb[g];
        #pragma unroll
        for (int c = 0; c < 8; ++c) {
            float xv = xp[(size_t)(g * 8 + c) * HWC];
            float v  = xv * cw[b * 64 + g * 8 + c];
            xr[c] = xv; xc[c] = v; dot += sw[g * 8 + c] * v;
        }
        const float sp  = sigm(dot);
        const float thr = rsum[b * 8 + g] * inv;
        #pragma unroll
        for (int c = 0; c < 8; ++c) {
            const float resp = sigm(xc[c] * sp);
            const float mk   = resp > thr ? 1.0f : resp;
            op[(size_t)(g * 8 + c) * HWC] = xr[c] * mk;
        }
    }
}

// global gate: mean of sigmoid(MLP(gb)) over all pixels
__global__ __launch_bounds__(256)
void gate_kernel(const float* __restrict__ gb,
                 const float* __restrict__ w1, const float* __restrict__ b1,
                 const float* __restrict__ w2, const float* __restrict__ b2,
                 float* __restrict__ gsum)
{
    const int b = blockIdx.y;
    const int s = blockIdx.x * 256 + threadIdx.x;
    const float* gp = gb + (size_t)b * 64 * HWC + s;
    float v[64];
    #pragma unroll
    for (int c = 0; c < 64; ++c) v[c] = gp[(size_t)c * HWC];
    float acc = b2[0];
    for (int j = 0; j < 32; ++j) {
        float a = b1[j];
        #pragma unroll
        for (int c = 0; c < 64; ++c) a += w1[j * 64 + c] * v[c];
        acc += w2[j] * (a > 0.0f ? a : 0.0f);
    }
    const float g = sigm(acc);
    const float r = waveRed(g);
    if ((threadIdx.x & 31) == 0) atomicAdd(gsum, r);
}

__device__ __forceinline__ float dw3x3(const float* __restrict__ in,
                                       const float* __restrict__ w9, int hh, int wc)
{
    float acc = 0.0f;
    #pragma unroll
    for (int ky = 0; ky < 3; ++ky) {
        const int y = hh + ky - 1;
        if ((unsigned)y >= (unsigned)HH) continue;
        #pragma unroll
        for (int kx = 0; kx < 3; ++kx) {
            const int x = wc + kx - 1;
            if ((unsigned)x >= (unsigned)WW) continue;
            acc += w9[ky * 3 + kx] * in[y * WW + x];
        }
    }
    return acc;
}

// depthwise conv for v channels (128..191) -> materialize v
__global__ __launch_bounds__(256)
void dw_v_kernel(const float* __restrict__ qkv, const float* __restrict__ dww,
                 float* __restrict__ vout)
{
    const int b = blockIdx.z, cy = blockIdx.y;
    const int s = blockIdx.x * 256 + threadIdx.x;
    const int hh = s >> 8, wc = s & 255;
    const int ch = 128 + cy;
    const float r = dw3x3(qkv + ((size_t)b * 192 + ch) * HWC, dww + ch * 9, hh, wc);
    vout[((size_t)b * 64 + cy) * HWC + s] = r;
}

// depthwise conv for q,k channels; accumulate sum-of-squares and 8x8 gram
// (L2 normalization of q,k factors out of the gram -> no q/k materialization)
__global__ __launch_bounds__(256)
void dw_qk_kernel(const float* __restrict__ qkv, const float* __restrict__ dww,
                  float* __restrict__ qss, float* __restrict__ kss,
                  float* __restrict__ G)
{
    const int b = blockIdx.z, h = blockIdx.y;
    const int s = blockIdx.x * 256 + threadIdx.x;
    const int hh = s >> 8, wc = s & 255;
    float qv[8], kv[8];
    #pragma unroll
    for (int c = 0; c < 8; ++c) {
        const int chq = h * 8 + c, chk = 64 + h * 8 + c;
        qv[c] = dw3x3(qkv + ((size_t)b * 192 + chq) * HWC, dww + chq * 9, hh, wc);
        kv[c] = dw3x3(qkv + ((size_t)b * 192 + chk) * HWC, dww + chk * 9, hh, wc);
    }
    const int lane = threadIdx.x & 31;
    #pragma unroll
    for (int c = 0; c < 8; ++c) {
        float r = waveRed(qv[c] * qv[c]);
        if (lane == 0) atomicAdd(&qss[b * 64 + h * 8 + c], r);
        r = waveRed(kv[c] * kv[c]);
        if (lane == 0) atomicAdd(&kss[b * 64 + h * 8 + c], r);
    }
    #pragma unroll
    for (int c = 0; c < 8; ++c)
        #pragma unroll
        for (int d = 0; d < 8; ++d) {
            const float r = waveRed(qv[c] * kv[d]);
            if (lane == 0) atomicAdd(&G[(((size_t)b * 8 + h) * 8 + c) * 8 + d], r);
        }
}

// single block: normalize gram, temperature, dynamic top-k rank mask, softmax,
// then fold attn + scales + proj_out into per-batch effective weight Weff[b].
__global__ __launch_bounds__(256)
void attn_kernel(const float* __restrict__ qss, const float* __restrict__ kss,
                 const float* __restrict__ G, const float* __restrict__ gsum,
                 const float* __restrict__ temp, const float* __restrict__ scales,
                 const float* __restrict__ wout, float* __restrict__ weff)
{
    __shared__ float attn_s[32 * 64];
    const int t = threadIdx.x;
    const float gmean = gsum[0] / (float)(BB * HWC);
    float kd = floorf(8.0f * gmean);
    kd = kd < 1.0f ? 1.0f : (kd > 8.0f ? 8.0f : kd);
    const int kdi = (int)kd;
    {
        const int bh = t >> 3, c = t & 7;
        const int b = bh >> 3, h = bh & 7;
        float nq = sqrtf(qss[b * 64 + h * 8 + c]); nq = nq > 1e-12f ? nq : 1e-12f;
        float arow[8];
        #pragma unroll
        for (int d = 0; d < 8; ++d) {
            float nk = sqrtf(kss[b * 64 + h * 8 + d]); nk = nk > 1e-12f ? nk : 1e-12f;
            arow[d] = G[((size_t)bh * 8 + c) * 8 + d] / (nq * nk) * temp[h];
        }
        float m = -3.4e38f;
        bool keep[8];
        #pragma unroll
        for (int d = 0; d < 8; ++d) {
            int rank = 0;
            #pragma unroll
            for (int j = 0; j < 8; ++j)
                if (arow[j] > arow[d] || (arow[j] == arow[d] && j < d)) rank++;
            keep[d] = rank < kdi;
            if (keep[d] && arow[d] > m) m = arow[d];
        }
        float e[8], sum = 0.0f;
        #pragma unroll
        for (int d = 0; d < 8; ++d) { e[d] = keep[d] ? __expf(arow[d] - m) : 0.0f; sum += e[d]; }
        #pragma unroll
        for (int d = 0; d < 8; ++d) attn_s[bh * 64 + c * 8 + d] = e[d] / sum;
    }
    __syncthreads();
    const float ssum = scales[0] + scales[1] + scales[2] + scales[3];
    for (int p = t; p < 512; p += 256) {
        const int b = p >> 7, o = p & 127;
        #pragma unroll
        for (int hh = 0; hh < 8; ++hh)
            #pragma unroll
            for (int d = 0; d < 8; ++d) {
                float a = 0.0f;
                #pragma unroll
                for (int c = 0; c < 8; ++c)
                    a += wout[o * 128 + hh * 8 + c] * attn_s[(b * 8 + hh) * 64 + c * 8 + d];
                weff[((size_t)b * 128 + o) * 128 + hh * 8 + d] = ssum * a;
            }
        for (int i = 64; i < 128; ++i)
            weff[((size_t)b * 128 + o) * 128 + i] = wout[o * 128 + i];
    }
}

// ---------------------------------------------------------------------------
// Launch
// ---------------------------------------------------------------------------
extern "C" void kernel_launch(void* const* d_in, const int* in_sizes, int n_in,
                              void* d_out, int out_size, void* d_ws, size_t ws_size,
                              hipStream_t stream)
{
    const float* x     = (const float*)d_in[0];
    const float* ipw   = (const float*)d_in[1];
    const float* prew  = (const float*)d_in[2];
    const float* preb  = (const float*)d_in[3];
    const float* cw1   = (const float*)d_in[4];
    const float* cb1   = (const float*)d_in[5];
    const float* cw2   = (const float*)d_in[6];
    const float* cb2   = (const float*)d_in[7];
    const float* csw   = (const float*)d_in[8];
    const float* csb   = (const float*)d_in[9];
    const float* postw = (const float*)d_in[10];
    const float* postb = (const float*)d_in[11];
    const float* qkvw  = (const float*)d_in[12];
    const float* dww   = (const float*)d_in[13];
    const float* gw1   = (const float*)d_in[14];
    const float* gb1   = (const float*)d_in[15];
    const float* gw2   = (const float*)d_in[16];
    const float* gb2   = (const float*)d_in[17];
    const float* temp  = (const float*)d_in[18];
    const float* scl   = (const float*)d_in[19];
    const float* poutw = (const float*)d_in[20];
    float* out = (float*)d_out;

    float* ws_f = (float*)d_ws;
    const size_t NBUF = (size_t)BB * 64 * HWC;       // 16,777,216 floats = 64 MB

    float* stats  = ws_f;                            // [0, 4096) small stats
    float* pooled = stats + 0;                       // 256
    float* cwv    = stats + 256;                     // 256
    float* rsum   = stats + 512;                     // 32
    float* gsum   = stats + 544;                     // 1
    float* qss    = stats + 576;                     // 256
    float* kss    = stats + 832;                     // 256
    float* G      = stats + 1088;                    // 2048
    float* weff   = stats + 4096;                    // 65536

    float* LB0   = ws_f + 131072;                    // 64 MB  local branch (residual)
    float* GB    = LB0 + NBUF;                       // 64 MB  global branch
    float* VBUF  = GB + NBUF;                        // 64 MB  v after dwconv
    float* QKV   = VBUF + NBUF;                      // 192 MB qkv (freed after dw)
    float* X1    = QKV;                              // alias: lgce pre output
    float* LBOUT = QKV + NBUF;                       // alias: lgce final output
    float* XM    = GB;                               // alias: masked activations

    const dim3 blk(256);

    zero_stats_kernel<<<16, blk, 0, stream>>>(stats);

    // input_proj: x[4,128,HW] -> lb0 (ch 0..63), gb (ch 64..127)
    gemm_bf16_kernel<128, 128, 128, 8><<<dim3(HWC / 128, BB), blk, 0, stream>>>(
        ipw, 0, x, nullptr, 128, nullptr, nullptr, LB0, GB, 64);

    // qkv 1x1: gb -> qkv[4,192,HW]
    gemm_bf16_kernel<192, 64, 64, 6><<<dim3(HWC / 64, BB), blk, 0, stream>>>(
        qkvw, 0, GB, nullptr, 64, nullptr, nullptr, QKV, QKV, 192);

    // global gate mean
    gate_kernel<<<dim3(HWC / 256, BB), blk, 0, stream>>>(GB, gw1, gb1, gw2, gb2, gsum);

    // depthwise 3x3: v -> VBUF, q/k -> reductions (qss, kss, gram)
    dw_v_kernel<<<dim3(HWC / 256, 64, BB), blk, 0, stream>>>(QKV, dww, VBUF);
    dw_qk_kernel<<<dim3(HWC / 256, 8, BB), blk, 0, stream>>>(QKV, dww, qss, kss, G);

    // LGCE pre conv (QKV region is free now; X1 aliases it)
    gemm_bf16_kernel<64, 64, 256, 8><<<dim3(HWC / 256, BB), blk, 0, stream>>>(
        prew, 0, LB0, nullptr, 64, preb, nullptr, X1, X1, 64);

    pool_kernel<<<dim3(64, BB), blk, 0, stream>>>(X1, pooled);
    mid_kernel<<<1, blk, 0, stream>>>(pooled, cw1, cb1, cw2, cb2, cwv, 1.0f / (float)HWC);
    cbam1_kernel<<<dim3(HWC / 256, BB), blk, 0, stream>>>(X1, cwv, csw, csb, rsum);
    mask_kernel<<<dim3(HWC / 256, BB), blk, 0, stream>>>(X1, cwv, csw, csb, rsum, XM);

    // LGCE post conv + bias + residual -> LBOUT
    gemm_bf16_kernel<64, 64, 256, 8><<<dim3(HWC / 256, BB), blk, 0, stream>>>(
        postw, 0, XM, nullptr, 64, postb, LB0, LBOUT, LBOUT, 64);

    // attn (8x8/head) + top-k + softmax folded with proj_out into Weff[b]
    attn_kernel<<<1, blk, 0, stream>>>(qss, kss, G, gsum, temp, scl, poutw, weff);

    // final: out = Weff[b] @ concat(v, lb)  (per-batch weights)
    gemm_bf16_kernel<128, 128, 128, 8><<<dim3(HWC / 128, BB), blk, 0, stream>>>(
        weff, (size_t)128 * 128, VBUF, LBOUT, 64, nullptr, nullptr, out, out, 128);
}
